// CopyDecoder_13357348290852
// MI455X (gfx1250) — matrix operand order repaired and checked by reading, not compiled, not run on hardware
//
#include <hip/hip_runtime.h>

// CopyDecoder for MI455X (gfx1250, wave32, WMMA bf16 16x16x32).
// Roofline: ~550 MB HBM traffic (~24us @ 23.3 TB/s), ~145 GFLOP -> bf16 WMMA
// with f32 accumulation. fp32->bf16 conversion done with one v_perm_b32 per
// element pair; reused operands (state, Wc_w) are pre-packed once into
// WMMA-fragment-layout bf16 so the hot GEMM loops do pure b128 loads + wmma.

typedef __attribute__((ext_vector_type(16))) __bf16 v16bf;
typedef __attribute__((ext_vector_type(8)))  float v8f;
typedef __attribute__((ext_vector_type(4)))  float f32x4;
typedef __attribute__((ext_vector_type(4)))  unsigned int u32x4;

namespace {

constexpr int B = 64, S = 512, V = 50257, E = 512, H = 1024, OOV = 12;
constexpr int H2 = 2 * H;            // 2048
constexpr int KGI = E + H2;          // 2560
constexpr int OUTV = V + OOV;        // 50269
constexpr int NT_G = (V + 15) / 16;  // 3142 N-tiles for score_g

// workspace layout (4-byte words)
constexpr size_t WS_SPACK  = 0;                              // 4*32*32*8 state bf16 frags
constexpr size_t WS_WCPACK = WS_SPACK + (size_t)4*32*32*8;   // 64*64*32*8 Wc bf16 frags
constexpr size_t WS_PREV   = WS_WCPACK + (size_t)64*64*32*8; // B*H   eff prev_state
constexpr size_t WS_WTD    = WS_PREV + (size_t)B * H;        // B*H2  eff weighted
constexpr size_t WS_SCG    = WS_WTD  + (size_t)B * H2;       // B*V   score_g
constexpr size_t WS_SCC    = WS_SCG  + (size_t)B * V;        // B*S   score_c
constexpr size_t WS_MX     = WS_SCC  + (size_t)B * S;        // B
constexpr size_t WS_SM     = WS_MX   + (size_t)B;            // B

union Frag {
  v16bf v;
  unsigned int d[8];
  u32x4 q[2];
};

// two fp32 -> packed bf16 pair in one v_perm_b32 (truncation)
static __device__ __forceinline__ unsigned int pkbf(float lo, float hi) {
  return __builtin_amdgcn_perm(__float_as_uint(hi), __float_as_uint(lo),
                               0x07060302u);
}

// 16 contiguous floats -> B fragment (element i: k = (lane/16)*16 + i)
static __device__ __forceinline__ void fill16(Frag& f, const f32x4* p) {
  f32x4 t0 = p[0], t1 = p[1], t2 = p[2], t3 = p[3];
  f.d[0] = pkbf(t0.x, t0.y); f.d[1] = pkbf(t0.z, t0.w);
  f.d[2] = pkbf(t1.x, t1.y); f.d[3] = pkbf(t1.z, t1.w);
  f.d[4] = pkbf(t2.x, t2.y); f.d[5] = pkbf(t2.z, t2.w);
  f.d[6] = pkbf(t3.x, t3.y); f.d[7] = pkbf(t3.z, t3.w);
}

// A fragment from a fp32 row at k-base: element i -> k = lh*16 + i
static __device__ __forceinline__ void fillA(Frag& f, const float* ap, int lh) {
  const f32x4* a0 = (const f32x4*)(ap + lh * 16);
  const f32x4* a1 = (const f32x4*)(ap + lh * 16 + 8);
  f32x4 t0 = a0[0], t1 = a0[1], t2 = a1[0], t3 = a1[1];
  f.d[0] = pkbf(t0.x, t0.y); f.d[1] = pkbf(t0.z, t0.w);
  f.d[2] = pkbf(t1.x, t1.y); f.d[3] = pkbf(t1.z, t1.w);
  f.d[4] = pkbf(t2.x, t2.y); f.d[5] = pkbf(t2.z, t2.w);
  f.d[6] = pkbf(t3.x, t3.y); f.d[7] = pkbf(t3.z, t3.w);
}

static __device__ __forceinline__ v8f wmma_bf16(Frag a, Frag b, v8f c) {
  return __builtin_amdgcn_wmma_f32_16x16x32_bf16(false, a.v, false, b.v,
                                                 (short)0, c, false, false);
}

static __device__ __forceinline__ float wave_sum(float v) {
#pragma unroll
  for (int m = 16; m >= 1; m >>= 1) v += __shfl_xor(v, m, 32);
  return v;
}
static __device__ __forceinline__ float half_sum(float v) {
#pragma unroll
  for (int m = 8; m >= 1; m >>= 1) v += __shfl_xor(v, m, 32);
  return v;
}
static __device__ __forceinline__ float sigmoidf(float x) {
  return 1.0f / (1.0f + expf(-x));
}

// ---- order==0 selector: eff_weighted = order ? weighted : 0 ---------------
__global__ void k_weighted_sel(const float* __restrict__ w,
                               const int* __restrict__ order,
                               float* __restrict__ dst) {
  int i = blockIdx.x * blockDim.x + threadIdx.x;
  if (i < B * H2) dst[i] = (*order == 0) ? 0.0f : w[i];
}

// ---- eff_prev = order ? prev : encoded[:, -1] @ Ws_w^T + Ws_b (wave GEMV) -
__global__ void k_prev_sel(const float* __restrict__ prev,
                           const float* __restrict__ encoded,
                           const float* __restrict__ Ws_w,
                           const float* __restrict__ Ws_b,
                           const int* __restrict__ order,
                           float* __restrict__ dst) {
  int w = blockIdx.x * 8 + threadIdx.x / 32;  // one wave per (b,h)
  int lane = threadIdx.x & 31;
  int b = w >> 10, h = w & (H - 1);
  if (*order != 0) {
    if (lane == 0) dst[w] = prev[w];
    return;
  }
  const float* er = encoded + (size_t)(b * S + (S - 1)) * H2;
  const float* wr = Ws_w + (size_t)h * H2;
  float s = 0.f;
  for (int k = lane; k < H2; k += 32) s = fmaf(er[k], wr[k], s);
  s = wave_sum(s);
  if (lane == 0) dst[w] = s + Ws_b[h];
}

// ---- pre-pack Wc_w into bf16 B-fragment layout (done once per call) -------
__global__ void k_wc_pack(const float* __restrict__ Wc_w,
                          unsigned int* __restrict__ wcpack) {
  int w = blockIdx.x * 8 + threadIdx.x / 32;  // (ntile*64 + kt32), 64*64 total
  int lane = threadIdx.x & 31;
  int ntile = w >> 6, kt = (w & 63) * 32;
  int ln = lane & 15, lh = lane >> 4;
  Frag f;
  fill16(f, (const f32x4*)(Wc_w + (size_t)(ntile * 16 + ln) * H2 + kt + lh * 16));
  u32x4* dst = (u32x4*)(wcpack + ((size_t)w * 32 + lane) * 8);
  dst[0] = f.q[0]; dst[1] = f.q[1];
}

// ---- fused GRU cell: one wave per (b, j) computes all 6 dots + gates ------
__global__ void k_gru(const int* __restrict__ input_idx,
                      const float* __restrict__ embed,
                      const float* __restrict__ wtd,
                      const float* __restrict__ prev,
                      const float* __restrict__ wi,
                      const float* __restrict__ wh,
                      const float* __restrict__ bi,
                      const float* __restrict__ bh,
                      float* __restrict__ state_out) {
  int w = blockIdx.x * 8 + threadIdx.x / 32;
  int lane = threadIdx.x & 31;
  int b = w >> 10, j = w & (H - 1);
  const float* xr0 = embed + (size_t)input_idx[b] * E;
  const float* xr1 = wtd + (size_t)b * H2;
  const float* wir = wi + (size_t)j * KGI;
  const float* whr = wh + (size_t)j * H;
  const float* pr = prev + (size_t)b * H;
  float sir = 0, siz = 0, sin_ = 0, shr = 0, shz = 0, shn = 0;
  for (int k = lane; k < KGI; k += 32) {
    float xv = (k < E) ? xr0[k] : xr1[k - E];
    sir  = fmaf(xv, wir[k], sir);
    siz  = fmaf(xv, wir[(size_t)H * KGI + k], siz);
    sin_ = fmaf(xv, wir[(size_t)2 * H * KGI + k], sin_);
  }
  for (int k = lane; k < H; k += 32) {
    float pv = pr[k];
    shr = fmaf(pv, whr[k], shr);
    shz = fmaf(pv, whr[(size_t)H * H + k], shz);
    shn = fmaf(pv, whr[(size_t)2 * H * H + k], shn);
  }
  sir = wave_sum(sir); siz = wave_sum(siz); sin_ = wave_sum(sin_);
  shr = wave_sum(shr); shz = wave_sum(shz); shn = wave_sum(shn);
  if (lane == 0) {
    float r = sigmoidf(sir + bi[j] + shr + bh[j]);
    float z = sigmoidf(siz + bi[H + j] + shz + bh[H + j]);
    float n = tanhf(sin_ + bi[2 * H + j] + r * (shn + bh[2 * H + j]));
    state_out[(size_t)b * H + j] = (1.f - z) * n + z * pr[j];
  }
}

// ---- pre-pack state into bf16 A-fragment layout ---------------------------
__global__ void k_state_pack(const float* __restrict__ state,
                             unsigned int* __restrict__ spack) {
  int w = blockIdx.x * 8 + threadIdx.x / 32;  // (mt*32 + kt32), 128 total
  int lane = threadIdx.x & 31;
  if (w >= 4 * (H / 32)) return;
  int mt = w >> 5, kt = (w & 31) * 32;
  int ln = lane & 15, lh = lane >> 4;
  Frag f;
  fillA(f, state + (size_t)(mt * 16 + ln) * H + kt, lh);
  u32x4* dst = (u32x4*)(spack + ((size_t)w * 32 + lane) * 8);
  dst[0] = f.q[0]; dst[1] = f.q[1];
}

// ---- score_g = state @ Wg^T + b : one wave per 16-col tile, 4 M-tiles -----
// Wg (206 MB) is streamed exactly once (HBM-bound); A frags are pure loads.
__global__ void k_score_g(const unsigned int* __restrict__ spack,
                          const float* __restrict__ Wg_w,
                          const float* __restrict__ Wg_b,
                          float* __restrict__ scg) {
  int wave = threadIdx.x / 32;
  int lane = threadIdx.x & 31;
  int ntile = blockIdx.x * 8 + wave;
  if (ntile >= NT_G) return;  // whole-wave exit keeps EXEC all-1s for WMMA
  int ln = lane & 15, lh = lane >> 4;
  int n = ntile * 16 + ln;
  int nclamp = n < V ? n : V - 1;
  // per-mt stride in u32x4 units: 32 kt-slots * 32 lanes * 2
  const u32x4* sp = (const u32x4*)spack + (size_t)lane * 2;
  v8f acc0 = {}, acc1 = {}, acc2 = {}, acc3 = {};
  for (int kt32 = 0; kt32 < H / 32; ++kt32) {
    const float* wp = Wg_w + (size_t)nclamp * H + kt32 * 32 + lh * 16;
    if (kt32 + 1 < H / 32) __builtin_prefetch(wp + 32, 0, 3);
    // load all 4 A fragments as one clause, then convert B, then 4 WMMAs
    const u32x4* s0 = sp + (size_t)kt32 * 64;
    Frag a0, a1, a2, a3;
    a0.q[0] = s0[0];        a0.q[1] = s0[1];
    a1.q[0] = s0[2048];     a1.q[1] = s0[2049];
    a2.q[0] = s0[4096];     a2.q[1] = s0[4097];
    a3.q[0] = s0[6144];     a3.q[1] = s0[6145];
    Frag bf;
    fill16(bf, (const f32x4*)wp);
    acc0 = wmma_bf16(a0, bf, acc0);
    acc1 = wmma_bf16(a1, bf, acc1);
    acc2 = wmma_bf16(a2, bf, acc2);
    acc3 = wmma_bf16(a3, bf, acc3);
  }
  if (n < V) {
    float bias = Wg_b[n];
    v8f acc[4] = {acc0, acc1, acc2, acc3};
#pragma unroll
    for (int mt = 0; mt < 4; ++mt)
#pragma unroll
      for (int r = 0; r < 8; ++r)
        scg[(size_t)(mt * 16 + r + 8 * lh) * V + n] = acc[mt][r] + bias;
  }
}

// ---- fused sc = tanh(encoded@Wc^T + b), score_c = tanh(sc . state) --------
// Workgroup = 16 waves, 64 rows x full N=1024; `sc` lives only in registers.
__global__ void __launch_bounds__(512) k_score_c(
    const float* __restrict__ encoded, const unsigned int* __restrict__ wcpack,
    const float* __restrict__ Wc_b, const float* __restrict__ state,
    const int* __restrict__ eidx, float* __restrict__ scc_out) {
  __shared__ float scc[64];
  int tid = threadIdx.x;
  int wave = tid / 32, lane = tid & 31;
  int ln = lane & 15, lh = lane >> 4;
  int row0 = blockIdx.x * 64;  // 64 flattened (b,s) rows, all in one b
  int b = row0 >> 9;
  if (tid < 64) scc[tid] = 0.f;
  __syncthreads();
  const u32x4* wq = (const u32x4*)wcpack + (size_t)lane * 2;
  v8f acc[4][4] = {};
  for (int kt32 = 0; kt32 < H2 / 32; ++kt32) {
    Frag bf[4];
#pragma unroll
    for (int nt = 0; nt < 4; ++nt) {
      const u32x4* w0 = wq + ((size_t)((wave * 4 + nt) * 64 + kt32)) * 64;
      bf[nt].q[0] = w0[0]; bf[nt].q[1] = w0[1];
    }
#pragma unroll
    for (int mt = 0; mt < 4; ++mt) {
      Frag af;
      fillA(af, encoded + (size_t)(row0 + mt * 16 + ln) * H2 + kt32 * 32, lh);
#pragma unroll
      for (int nt = 0; nt < 4; ++nt)
        acc[mt][nt] = wmma_bf16(af, bf[nt], acc[mt][nt]);
    }
  }
  const float* st = state + (size_t)b * H;
#pragma unroll
  for (int mt = 0; mt < 4; ++mt) {
#pragma unroll
    for (int r = 0; r < 8; ++r) {
      float p = 0.f;
#pragma unroll
      for (int nt = 0; nt < 4; ++nt) {
        int nn = wave * 64 + nt * 16 + ln;
        p += tanhf(acc[mt][nt][r] + Wc_b[nn]) * st[nn];
      }
      p = half_sum(p);  // sum over the 16 n-lanes
      if (ln == 0) atomicAdd(&scc[mt * 16 + r + 8 * lh], p);
    }
  }
  __syncthreads();
  if (tid < 64) {
    int s = (row0 + tid) & (S - 1);
    float v = tanhf(scc[tid]);
    if (eidx[(size_t)b * S + s] == 0) v -= 1000.f;
    scc_out[(size_t)b * S + s] = v;
  }
}

// ---- softmax stats over concat(score_g, score_c) per batch row ------------
__global__ void k_softmax_stats(const float* __restrict__ scg,
                                const float* __restrict__ scc,
                                float* __restrict__ mx_out,
                                float* __restrict__ sm_out) {
  __shared__ float red[256];
  int b = blockIdx.x, tid = threadIdx.x;
  const float* g = scg + (size_t)b * V;
  const float* c = scc + (size_t)b * S;
  float m = -3.4e38f;
  for (int v = tid; v < V; v += 256) m = fmaxf(m, g[v]);
  for (int s = tid; s < S; s += 256) m = fmaxf(m, c[s]);
  red[tid] = m; __syncthreads();
  for (int o = 128; o > 0; o >>= 1) {
    if (tid < o) red[tid] = fmaxf(red[tid], red[tid + o]);
    __syncthreads();
  }
  float mx = red[0]; __syncthreads();
  float s = 0.f;
  for (int v = tid; v < V; v += 256) s += expf(g[v] - mx);
  for (int i = tid; i < S; i += 256) s += expf(c[i] - mx);
  red[tid] = s; __syncthreads();
  for (int o = 128; o > 0; o >>= 1) {
    if (tid < o) red[tid] += red[tid + o];
    __syncthreads();
  }
  if (tid == 0) { mx_out[b] = mx; sm_out[b] = red[0]; }
}

// ---- out[:, :V] = prob_g, out[:, V:] = 1e-4 -------------------------------
__global__ void k_prob_g(const float* __restrict__ scg,
                         const float* __restrict__ mx,
                         const float* __restrict__ sm,
                         float* __restrict__ out) {
  int v = blockIdx.x * 256 + threadIdx.x;
  int b = blockIdx.y;
  if (v >= OUTV) return;
  float r = (v < V) ? expf(scg[(size_t)b * V + v] - mx[b]) / sm[b] : 1e-4f;
  out[(size_t)b * OUTV + v] = r;
}

// ---- scatter prob_c into out; sparse attention-weighted sum of encoded ----
__global__ void __launch_bounds__(512) k_scatter_attn(
    const float* __restrict__ scc, const int* __restrict__ eidx,
    const int* __restrict__ input_idx, const float* __restrict__ encoded,
    const float* __restrict__ mx, const float* __restrict__ sm,
    float* __restrict__ out, float* __restrict__ wout) {
  __shared__ float red[512];
  int b = blockIdx.x, tid = threadIdx.x;
  for (int i = tid; i < H2; i += 512) wout[(size_t)b * H2 + i] = 0.f;
  int s = tid;  // 512 threads == S
  int idx = eidx[(size_t)b * S + s];
  float pc = expf(scc[(size_t)b * S + s] - mx[b]) / sm[b];
  atomicAdd(&out[(size_t)b * OUTV + idx], pc);
  int match = (idx == input_idx[b]) ? 1 : 0;
  red[tid] = (float)match; __syncthreads();
  for (int o = 256; o > 0; o >>= 1) {
    if (tid < o) red[tid] += red[tid + o];
    __syncthreads();
  }
  float ssum = red[0];
  float wgt = (ssum > 1.f) ? 1.f / ssum : 1.f;
  if (match) {  // rare: random vocab indices -> ~0 matches per row
    float a = pc * wgt;
    const float* er = encoded + (size_t)(b * S + s) * H2;
    for (int d = 0; d < H2; ++d)
      atomicAdd(&wout[(size_t)b * H2 + d], a * er[d]);
  }
}

}  // namespace

extern "C" void kernel_launch(void* const* d_in, const int* in_sizes, int n_in,
                              void* d_out, int out_size, void* d_ws,
                              size_t ws_size, hipStream_t stream) {
  (void)in_sizes; (void)n_in; (void)out_size; (void)ws_size;
  const int*   input_idx = (const int*)d_in[0];
  const float* encoded   = (const float*)d_in[1];
  const int*   eidx      = (const int*)d_in[2];
  const float* prev      = (const float*)d_in[3];
  const float* weighted  = (const float*)d_in[4];
  const int*   order     = (const int*)d_in[5];
  const float* embed     = (const float*)d_in[6];
  const float* gru_wi    = (const float*)d_in[7];
  const float* gru_wh    = (const float*)d_in[8];
  const float* gru_bi    = (const float*)d_in[9];
  const float* gru_bh    = (const float*)d_in[10];
  const float* Ws_w      = (const float*)d_in[11];
  const float* Ws_b      = (const float*)d_in[12];
  const float* Wg_w      = (const float*)d_in[13];
  const float* Wg_b      = (const float*)d_in[14];
  const float* Wc_w      = (const float*)d_in[15];
  const float* Wc_b      = (const float*)d_in[16];

  unsigned int* wsw = (unsigned int*)d_ws;
  float* ws = (float*)d_ws;
  unsigned int* spack  = wsw + WS_SPACK;
  unsigned int* wcpack = wsw + WS_WCPACK;
  float* eff_prev = ws + WS_PREV;
  float* eff_wtd  = ws + WS_WTD;
  float* scg      = ws + WS_SCG;
  float* scc      = ws + WS_SCC;
  float* mx       = ws + WS_MX;
  float* sm       = ws + WS_SM;

  float* out0  = (float*)d_out;                 // B x OUTV
  float* state = out0 + (size_t)B * OUTV;       // B x H
  float* wout  = state + (size_t)B * H;         // B x H2

  k_weighted_sel<<<(B * H2 + 255) / 256, 256, 0, stream>>>(weighted, order, eff_wtd);
  k_prev_sel<<<(B * H) / 8, 256, 0, stream>>>(prev, encoded, Ws_w, Ws_b, order, eff_prev);
  k_wc_pack<<<(64 * 64) / 8, 256, 0, stream>>>(Wc_w, wcpack);
  k_gru<<<(B * H) / 8, 256, 0, stream>>>(input_idx, embed, eff_wtd, eff_prev,
                                         gru_wi, gru_wh, gru_bi, gru_bh, state);
  k_state_pack<<<16, 256, 0, stream>>>(state, spack);
  k_score_g<<<(NT_G + 7) / 8, 256, 0, stream>>>(spack, Wg_w, Wg_b, scg);
  k_score_c<<<(B * S) / 64, 512, 0, stream>>>(encoded, wcpack, Wc_b, state, eidx, scc);
  k_softmax_stats<<<B, 256, 0, stream>>>(scg, scc, mx, sm);
  k_prob_g<<<dim3((OUTV + 255) / 256, B), 256, 0, stream>>>(scg, mx, sm, out0);
  k_scatter_attn<<<B, 512, 0, stream>>>(scc, eidx, input_idx, encoded, mx, sm, out0, wout);
}
